// AGNNet_541165879486
// MI455X (gfx1250) — matrix operand order repaired and checked by reading, not compiled, not run on hardware
//
#include <hip/hip_runtime.h>

// ---------------------------------------------------------------------------
// AGNN forward for MI455X (gfx1250, wave32).
// GEMMs via v_wmma_f32_16x16x32_f16 with pre-packed f16 weight fragments;
// edge ops via f32 global atomics.
// ---------------------------------------------------------------------------

#define HID 64

typedef _Float16 v16h __attribute__((ext_vector_type(16)));
typedef float    v8f  __attribute__((ext_vector_type(8)));
typedef float    v4f  __attribute__((ext_vector_type(4)));

__device__ __forceinline__ float wave_reduce_add(float v) {
#pragma unroll
    for (int o = 16; o >= 1; o >>= 1) v += __shfl_xor(v, o, 32);
    return v;
}

// ---------------------------------------------------------------------------
// Repack fp32 weights W[K x Nc] (row-major) into f16 WMMA B-fragments.
// Fragment f = ks*NT + t holds a 32x16 B tile in ISA 7.12.2 layout:
//   element e of lane l:  col = t*16 + (l&15),  kk = ks*32 + ((l>=16)?8:0) + (e&7) + ((e>=8)?16:0)
//   stored at out[f*512 + l*16 + e]
// Out-of-range columns (ragged Nc, e.g. 40) are zero-filled here, so the GEMM
// inner loop is branch-free.
// ---------------------------------------------------------------------------
__global__ void agnn_repack_w(const float* __restrict__ W, _Float16* __restrict__ out,
                              int K, int Nc, int NT)
{
    const int idx   = blockIdx.x * blockDim.x + threadIdx.x;
    const int total = (K >> 5) * NT * 32 * 16;
    if (idx >= total) return;
    const int e    = idx & 15;
    const int lane = (idx >> 4) & 31;
    const int f    = idx >> 9;
    const int t    = f % NT;
    const int ks   = f / NT;
    const int col  = t * 16 + (lane & 15);
    const int kk   = ks * 32 + ((lane >> 4) << 3) + (e & 7) + ((e >> 3) << 4);
    const float w  = (col < Nc) ? W[(size_t)kk * Nc + col] : 0.0f;
    out[idx] = (_Float16)w;
}

// ---------------------------------------------------------------------------
// C[M x Nc] = act(A[M x K] @ W + bias). One wave per 16-row strip; the A
// fragment is loaded once per k-step and reused across all NT column tiles.
// A is fp32 row-major (ld = K), Wf is pre-packed f16 fragments, C ld = Nc.
// M multiple of 16 (100000 = 6250*16), K multiple of 32.
// ---------------------------------------------------------------------------
template <int NT>
__global__ void agnn_gemm_wmma(const float* __restrict__ A,
                               const v16h* __restrict__ Wf,
                               const float* __restrict__ bias,
                               float* __restrict__ C,
                               int M, int K, int Nc, int relu)
{
    const int lane = threadIdx.x & 31;
    const int wave = threadIdx.x >> 5;
    const int tm   = blockIdx.x * (blockDim.x >> 5) + wave;
    if (tm >= (M >> 4)) return;            // wave-uniform: EXEC all-1s below

    const int half = lane >> 4;            // 0: lanes 0-15, 1: lanes 16-31
    const int row  = tm * 16 + (lane & 15);

    v8f c[NT];
#pragma unroll
    for (int t = 0; t < NT; ++t) c[t] = (v8f){};

    const int ksteps = K >> 5;
    for (int ks = 0; ks < ksteps; ++ks) {
        // A fragment (16-bit A 16x32 layout, ISA 7.12.2):
        //   lanes 0-15:  elems 0..7 = K[0..7],  elems 8..15 = K[16..23]
        //   lanes16-31:  elems 0..7 = K[8..15], elems 8..15 = K[24..31]
        const float* arow = A + (size_t)row * K + ks * 32 + half * 8;
        const v4f a0 = *(const v4f*)(arow);
        const v4f a1 = *(const v4f*)(arow + 4);
        const v4f a2 = *(const v4f*)(arow + 16);
        const v4f a3 = *(const v4f*)(arow + 20);
        v16h a;
#pragma unroll
        for (int i = 0; i < 4; ++i) {
            a[i]      = (_Float16)a0[i];
            a[i + 4]  = (_Float16)a1[i];
            a[i + 8]  = (_Float16)a2[i];
            a[i + 12] = (_Float16)a3[i];
        }
#pragma unroll
        for (int t = 0; t < NT; ++t) {
            const v16h b = Wf[(ks * NT + t) * 32 + lane];
            c[t] = __builtin_amdgcn_wmma_f32_16x16x32_f16(false, a, false, b,
                                                          (short)0, c[t], false, false);
        }
    }

#pragma unroll
    for (int t = 0; t < NT; ++t) {
        const int col = t * 16 + (lane & 15);
        if (col < Nc) {
            const float bb = bias[col];
#pragma unroll
            for (int v = 0; v < 8; ++v) {
                const int m = tm * 16 + v + (half << 3);   // C/D layout (ISA 7.12.2)
                float val = c[t][v] + bb;
                if (relu) val = fmaxf(val, 0.0f);
                C[(size_t)m * Nc + col] = val;
            }
        }
    }
}

// ---------------------------------------------------------------------------
// Per-node fused reductions over h0 (one wave per node, 64 cols = 2 per lane):
//   delta = sum|h0|, ai = h0 . att_w[0:64], aj = h0 . att_w[64:128], hwp = h0 . wp
// ---------------------------------------------------------------------------
__global__ void agnn_node_stats(const float* __restrict__ h0,
                                const float* __restrict__ wp,
                                const float* __restrict__ att_w,
                                float* __restrict__ delta,
                                float* __restrict__ ai,
                                float* __restrict__ aj,
                                float* __restrict__ hwp,
                                int n)
{
    const int node = (int)((blockIdx.x * (long long)blockDim.x + threadIdx.x) >> 5);
    const int lane = threadIdx.x & 31;
    if (node >= n) return;
    const float* r = h0 + (size_t)node * HID;
    const float v0 = r[lane], v1 = r[lane + 32];
    float s_abs = fabsf(v0) + fabsf(v1);
    float s_ai  = v0 * att_w[lane]      + v1 * att_w[lane + 32];
    float s_aj  = v0 * att_w[64 + lane] + v1 * att_w[96 + lane];
    float s_wp  = v0 * wp[lane]         + v1 * wp[lane + 32];
    s_abs = wave_reduce_add(s_abs);
    s_ai  = wave_reduce_add(s_ai);
    s_aj  = wave_reduce_add(s_aj);
    s_wp  = wave_reduce_add(s_wp);
    if (lane == 0) {
        delta[node] = s_abs; ai[node] = s_ai; aj[node] = s_aj; hwp[node] = s_wp;
    }
}

// neigh_sum[dst] += delta[src]
__global__ void agnn_edge_neigh(const int* __restrict__ src, const int* __restrict__ dst,
                                const float* __restrict__ delta,
                                float* __restrict__ neigh, int Ecnt)
{
    const int e = blockIdx.x * blockDim.x + threadIdx.x;
    if (e >= Ecnt) return;
    atomicAdd(&neigh[dst[e]], delta[src[e]]);
}

// pi = sigmoid(h0@wp + neigh_sum)
__global__ void agnn_pi(const float* __restrict__ hwp, const float* __restrict__ neigh,
                        float* __restrict__ pi, int n)
{
    const int i = blockIdx.x * blockDim.x + threadIdx.x;
    if (i >= n) return;
    pi[i] = 1.0f / (1.0f + expf(-(hwp[i] + neigh[i])));
}

// e = leaky_relu(ai[dst] + aj[src] + pi[src]*w_p + att_b); expe = exp(e); denom[dst] += expe
__global__ void agnn_edge_att(const int* __restrict__ src, const int* __restrict__ dst,
                              const float* __restrict__ ai, const float* __restrict__ aj,
                              const float* __restrict__ pi,
                              const float* __restrict__ att_w,   // [2*HID] is w_p
                              const float* __restrict__ att_b,   // scalar
                              float* __restrict__ expe, float* __restrict__ denom, int Ecnt)
{
    const int e = blockIdx.x * blockDim.x + threadIdx.x;
    if (e >= Ecnt) return;
    const int s = src[e], d = dst[e];
    float v = ai[d] + aj[s] + pi[s] * att_w[2 * HID] + att_b[0];
    v = (v > 0.0f) ? v : 0.2f * v;
    const float x = expf(v);
    expe[e] = x;
    atomicAdd(&denom[d], x);
}

// alpha = expe / (denom[dst] + 1e-16)   (in place on expe)
__global__ void agnn_alpha(const int* __restrict__ dst, const float* __restrict__ denom,
                           float* __restrict__ expe, int Ecnt)
{
    const int e = blockIdx.x * blockDim.x + threadIdx.x;
    if (e >= Ecnt) return;
    expe[e] = expe[e] / (denom[dst[e]] + 1e-16f);
}

// acc[dst, c] += alpha[e] * hl[src, c]   (64 threads per edge, coalesced)
__global__ void agnn_edge_aggregate(const int* __restrict__ src, const int* __restrict__ dst,
                                    const float* __restrict__ alpha,
                                    const float* __restrict__ hl,
                                    float* __restrict__ acc, int Ecnt)
{
    const long long t = (long long)blockIdx.x * blockDim.x + threadIdx.x;
    if (t >= (long long)Ecnt * HID) return;
    const int e = (int)(t >> 6);
    const int c = (int)(t & 63);
    atomicAdd(&acc[(size_t)dst[e] * HID + c], alpha[e] * hl[(size_t)src[e] * HID + c]);
}

// h = relu(acc)
__global__ void agnn_relu(const float* __restrict__ acc, float* __restrict__ h, long long n)
{
    const long long i = (long long)blockIdx.x * blockDim.x + threadIdx.x;
    if (i >= n) return;
    h[i] = fmaxf(acc[i], 0.0f);
}

// ---------------------------------------------------------------------------

extern "C" void kernel_launch(void* const* d_in, const int* in_sizes, int n_in,
                              void* d_out, int out_size, void* d_ws, size_t ws_size,
                              hipStream_t stream)
{
    const int IN_C  = 128;
    const int OUT_C = 40;
    const int Ncnt  = in_sizes[0] / IN_C;   // 100000
    const int Ecnt  = in_sizes[1] / 2;      // 800000

    const float* x      = (const float*)d_in[0];
    const int*   eidx   = (const int*)  d_in[1];
    const float* W_in   = (const float*)d_in[2];
    const float* b_in   = (const float*)d_in[3];
    const float* wp     = (const float*)d_in[4];
    const float* att_w  = (const float*)d_in[5];
    const float* att_b  = (const float*)d_in[6];
    const float* W_out  = (const float*)d_in[7];
    const float* b_out  = (const float*)d_in[8];
    const float* Wl[3]  = {(const float*)d_in[9],  (const float*)d_in[11], (const float*)d_in[13]};
    const float* bl[3]  = {(const float*)d_in[10], (const float*)d_in[12], (const float*)d_in[14]};
    float* out = (float*)d_out;

    const int* src = eidx;
    const int* dst = eidx + Ecnt;

    // ---- workspace carve (256B aligned) ----
    char*  base = (char*)d_ws;
    size_t off  = 0;
    auto carve = [&](size_t bytes) -> void* {
        void* p = (void*)(base + off);
        off = (off + bytes + 255) & ~(size_t)255;
        return p;
    };
    const size_t hbytes = (size_t)Ncnt * HID * sizeof(float);
    float* h     = (float*)carve(hbytes);           // activations (h0, then per-layer h)
    float* hl    = (float*)carve(hbytes);           // h @ W_l + b_l
    float* acc   = (float*)carve(hbytes);           // segment-sum accumulator
    float* delta = (float*)carve(Ncnt * sizeof(float));
    float* ai    = (float*)carve(Ncnt * sizeof(float));
    float* aj    = (float*)carve(Ncnt * sizeof(float));
    float* hwp   = (float*)carve(Ncnt * sizeof(float));
    float* neigh = (float*)carve(Ncnt * sizeof(float));
    float* pi    = (float*)carve(Ncnt * sizeof(float));
    float* denom = (float*)carve(Ncnt * sizeof(float));
    float* expe  = (float*)carve((size_t)Ecnt * sizeof(float));
    // f16 weight-fragment buffers
    const int fin_elems  = (IN_C / 32) * 4 * 512;   // W_in:  4 ksteps * 4 ntiles
    const int fl_elems   = (HID  / 32) * 4 * 512;   // W_l:   2 ksteps * 4 ntiles
    const int fout_elems = (HID  / 32) * 3 * 512;   // W_out: 2 ksteps * 3 ntiles
    _Float16* wf_in  = (_Float16*)carve(fin_elems  * sizeof(_Float16));
    _Float16* wf_l0  = (_Float16*)carve(fl_elems   * sizeof(_Float16));
    _Float16* wf_l1  = (_Float16*)carve(fl_elems   * sizeof(_Float16));
    _Float16* wf_l2  = (_Float16*)carve(fl_elems   * sizeof(_Float16));
    _Float16* wf_out = (_Float16*)carve(fout_elems * sizeof(_Float16));
    _Float16* wf_l[3] = {wf_l0, wf_l1, wf_l2};
    (void)ws_size; (void)n_in; (void)out_size;

    const int BLK         = 256;
    const int wavesPerBlk = BLK / 32;
    const int gemmBlocks  = ((Ncnt / 16) + wavesPerBlk - 1) / wavesPerBlk;

    // 0) repack all weights into f16 WMMA fragments (tiny, once per launch)
    agnn_repack_w<<<(fin_elems + BLK - 1) / BLK, BLK, 0, stream>>>(W_in, wf_in, IN_C, HID, 4);
    for (int l = 0; l < 3; ++l)
        agnn_repack_w<<<(fl_elems + BLK - 1) / BLK, BLK, 0, stream>>>(Wl[l], wf_l[l], HID, HID, 4);
    agnn_repack_w<<<(fout_elems + BLK - 1) / BLK, BLK, 0, stream>>>(W_out, wf_out, HID, OUT_C, 3);

    // 1) h0 = relu(x @ W_in + b_in)      [100000 x 128 @ 128 x 64]
    agnn_gemm_wmma<4><<<gemmBlocks, BLK, 0, stream>>>(
        x, (const v16h*)wf_in, b_in, h, Ncnt, IN_C, HID, 1);

    // 2) per-node fused reductions over h0
    agnn_node_stats<<<(Ncnt * 32 + BLK - 1) / BLK, BLK, 0, stream>>>(
        h, wp, att_w, delta, ai, aj, hwp, Ncnt);

    // 3) neigh_sum / pi
    hipMemsetAsync(neigh, 0, Ncnt * sizeof(float), stream);
    hipMemsetAsync(denom, 0, Ncnt * sizeof(float), stream);
    agnn_edge_neigh<<<(Ecnt + BLK - 1) / BLK, BLK, 0, stream>>>(src, dst, delta, neigh, Ecnt);
    agnn_pi<<<(Ncnt + BLK - 1) / BLK, BLK, 0, stream>>>(hwp, neigh, pi, Ncnt);

    // 4) edge attention -> alpha
    agnn_edge_att<<<(Ecnt + BLK - 1) / BLK, BLK, 0, stream>>>(
        src, dst, ai, aj, pi, att_w, att_b, expe, denom, Ecnt);
    agnn_alpha<<<(Ecnt + BLK - 1) / BLK, BLK, 0, stream>>>(dst, denom, expe, Ecnt);

    // 5) three message-passing layers
    const long long aggThreads = (long long)Ecnt * HID;
    const int aggBlocks = (int)((aggThreads + BLK - 1) / BLK);
    const long long hElems = (long long)Ncnt * HID;
    const int hBlocks = (int)((hElems + BLK - 1) / BLK);
    for (int l = 0; l < 3; ++l) {
        agnn_gemm_wmma<4><<<gemmBlocks, BLK, 0, stream>>>(
            h, (const v16h*)wf_l[l], bl[l], hl, Ncnt, HID, HID, 0);
        hipMemsetAsync(acc, 0, hbytes, stream);
        agnn_edge_aggregate<<<aggBlocks, BLK, 0, stream>>>(src, dst, expe, hl, acc, Ecnt);
        agnn_relu<<<hBlocks, BLK, 0, stream>>>(acc, h, hElems);
    }

    // 6) out = h @ W_out + b_out         [100000 x 64 @ 64 x 40]
    agnn_gemm_wmma<3><<<gemmBlocks, BLK, 0, stream>>>(
        h, (const v16h*)wf_out, b_out, out, Ncnt, HID, OUT_C, 0);
}
